// BipartiteRGCN_9397388443819
// MI455X (gfx1250) — matrix operand order repaired and checked by reading, not compiled, not run on hardware
//
#include <hip/hip_runtime.h>

typedef __attribute__((ext_vector_type(16))) __bf16 v16bf;
typedef __attribute__((ext_vector_type(8)))  __bf16 v8bf;
typedef __attribute__((ext_vector_type(4)))  __bf16 v4bf;
typedef __attribute__((ext_vector_type(8)))  float  v8f;
typedef __attribute__((ext_vector_type(4)))  float  v4f;

constexpr int kRepos = 30000;
constexpr int kUsers = 70000;
constexpr int kN     = 100000;   // kRepos + kUsers
constexpr int kR     = 8;
constexpr int kE     = 1000000;
constexpr int kIn    = 8;
constexpr int kH     = 128;
constexpr int kKagg  = kR * kH;        // 1024
constexpr int kKtot  = kKagg + kH;     // 1152 = [8 relations | root]

// ---------------- workspace layout ----------------
// f32 region (offsets in floats)
constexpr size_t AGG1     = 0;
constexpr size_t SZ_AGG1  = (size_t)kUsers * kR * kH;   // 71,680,000
constexpr size_t CNT1     = AGG1 + SZ_AGG1;
constexpr size_t SZ_CNT1  = (size_t)kUsers * kR;        // 560,000
constexpr size_t AGG2     = CNT1 + SZ_CNT1;
constexpr size_t SZ_AGG2  = (size_t)kRepos * kR * kH;   // 30,720,000
constexpr size_t CNT2     = AGG2 + SZ_AGG2;
constexpr size_t SZ_CNT2  = (size_t)kRepos * kR;        // 240,000
constexpr size_t ZERO_TOT = CNT2 + SZ_CNT2;             // 103,200,000 (zeroed every call)
constexpr size_t H2OFF    = ZERO_TOT;
constexpr size_t SZ_H2    = (size_t)kRepos * kH;        // 3,840,000
constexpr size_t F32_TOT  = H2OFF + SZ_H2;
// bf16 region (offsets in __bf16 elements), base = ws + F32_TOT*4 bytes
constexpr size_t HB    = 0;                        // h  = relu(x@mlp_w+b), all N nodes
constexpr size_t SZ_HB = (size_t)kN * kH;
constexpr size_t H1B   = HB + SZ_HB;               // h1 after layer 1, all N nodes
constexpr size_t WT1   = H1B + SZ_HB;              // transposed [W1;root1]: (128, 1152)
constexpr size_t SZ_WT = (size_t)kH * kKtot;
constexpr size_t WT2   = WT1 + SZ_WT;

// ---------------- kernels ----------------

__global__ void k_zero(float* __restrict__ p, size_t n4) {
  size_t i = (size_t)blockIdx.x * blockDim.x + threadIdx.x;
  if (i < n4) ((v4f*)p)[i] = v4f{0.f, 0.f, 0.f, 0.f};
}

// Build Wt[n][k] = (k<1024 ? w[k/128][k%128][n] : root[k-1024][n]) in bf16.
__global__ void k_prep_w(const float* __restrict__ w1, const float* __restrict__ r1,
                         const float* __restrict__ w2, const float* __restrict__ r2,
                         __bf16* __restrict__ wt1, __bf16* __restrict__ wt2) {
  int tid = blockIdx.x * blockDim.x + threadIdx.x;
  int total = kH * kKtot;
  if (tid >= 2 * total) return;
  const float* w  = (tid < total) ? w1  : w2;
  const float* rt = (tid < total) ? r1  : r2;
  __bf16*      o  = (tid < total) ? wt1 : wt2;
  int t = tid % total;
  int n = t / kKtot;
  int k = t % kKtot;
  float v = (k < kKagg)
              ? w[(size_t)(k >> 7) * kH * kH + (size_t)(k & 127) * kH + n]
              : rt[(size_t)(k - kKagg) * kH + n];
  o[(size_t)n * kKtot + k] = (__bf16)v;
}

// h[n][j] = relu(sum_i x[n][i] * mlp_w[i][j] + mlp_b[j]) stored as bf16
__global__ void k_mlp(const float* __restrict__ x, const float* __restrict__ w,
                      const float* __restrict__ b, __bf16* __restrict__ h) {
  int tid = blockIdx.x * blockDim.x + threadIdx.x;
  if (tid >= kN * kH) return;
  int n = tid >> 7, j = tid & 127;
  float acc = b[j];
#pragma unroll
  for (int i = 0; i < kIn; ++i) acc += x[(size_t)n * kIn + i] * w[i * kH + j];
  h[tid] = (__bf16)fmaxf(acc, 0.f);
}

// One wave per edge: atomically add the 128-wide bf16 message (as f32) into
// agg[(dst*R+type)*128 + k], and count the edge.
__global__ void k_scatter(const __bf16* __restrict__ msg, const int* __restrict__ esrc,
                          const int* __restrict__ edst, const int* __restrict__ etype,
                          int srcAdd, float* __restrict__ agg, float* __restrict__ cnt) {
  int wv = threadIdx.x >> 5, lane = threadIdx.x & 31;
  int e = blockIdx.x * 8 + wv;
  if (e >= kE) return;
  int src    = esrc[e] + srcAdd;
  int bucket = edst[e] * kR + etype[e];
  const __bf16* m = msg + (size_t)src * kH + lane * 4;
  float*        a = agg + (size_t)bucket * kH + lane * 4;
  v4bf v = *(const v4bf*)m;
  unsafeAtomicAdd(a + 0, (float)v[0]);
  unsafeAtomicAdd(a + 1, (float)v[1]);
  unsafeAtomicAdd(a + 2, (float)v[2]);
  unsafeAtomicAdd(a + 3, (float)v[3]);
  if (lane == 0) unsafeAtomicAdd(cnt + bucket, 1.0f);
}

// Fused RGCN layer GEMM:  out = relu([mean(agg) | hroot] @ Wt^T + bias)
// One WAVE owns a 16-row x 128-col output tile (8 x v8f accumulators), so the
// A-fragment (agg mean data, the big operand) is loaded exactly once; the 8
// column tiles re-read the small L2-resident weight matrix.
// A (16x32 bf16): lanes 0-15 row M=lane, K kb..kb+15;
//                 lanes 16-31 row M=lane-16, K kb+16..kb+31.
// B (32x16 bf16): lane -> col N=lane&15; 16 contiguous K at kb + (lane<16?0:16)
//                 (Wt stored N-major, K contiguous).
// C/D f32: vgpr v -> row m0 + v + (lane<16?0:8), col n0 + (lane&15).
template <bool OUT_BF16>
__global__ __launch_bounds__(256) void k_gemm(
    const float* __restrict__ agg, const float* __restrict__ cnt,
    const __bf16* __restrict__ hroot, const __bf16* __restrict__ wt,
    const float* __restrict__ bias, __bf16* __restrict__ obf,
    float* __restrict__ of32, int numTiles, int aggRowStart) {
  int wtile = blockIdx.x * 8 + (threadIdx.x >> 5);  // 16-row tile id (wave-uniform)
  if (wtile >= numTiles) return;                    // wave-uniform exit
  int lane = threadIdx.x & 31;
  int l    = lane & 15;
  bool hi  = lane >= 16;
  int m0   = wtile * 16;
  int row  = m0 + l;

  v8f acc[8];
#pragma unroll
  for (int n = 0; n < 8; ++n) acc[n] = {};

  const __bf16* wbase = wt + (size_t)l * kKtot + (hi ? 16 : 0);

  if (m0 >= aggRowStart) {  // wave-uniform: EXEC stays all-ones
    int ar = row - aggRowStart;
    const float* arow = agg + (size_t)ar * kKagg;
    float sc[kR];
#pragma unroll
    for (int r = 0; r < kR; ++r) sc[r] = 1.0f / fmaxf(cnt[(size_t)ar * kR + r], 1.0f);
#pragma unroll 1
    for (int kb = 0; kb < kKagg; kb += 32) {
      int koff = kb + (hi ? 16 : 0);
      v4f f0 = *(const v4f*)(arow + koff);
      v4f f1 = *(const v4f*)(arow + koff + 4);
      v4f f2 = *(const v4f*)(arow + koff + 8);
      v4f f3 = *(const v4f*)(arow + koff + 12);
      float s = sc[kb >> 7];
      v16bf a;
#pragma unroll
      for (int i = 0; i < 4; ++i) {
        a[i]      = (__bf16)(f0[i] * s);
        a[4 + i]  = (__bf16)(f1[i] * s);
        a[8 + i]  = (__bf16)(f2[i] * s);
        a[12 + i] = (__bf16)(f3[i] * s);
      }
#pragma unroll
      for (int n = 0; n < 8; ++n) {
        v16bf b = *(const v16bf*)(wbase + (size_t)n * 16 * kKtot + kb);
        acc[n] = __builtin_amdgcn_wmma_f32_16x16x32_bf16(false, a, false, b,
                                                         (short)0, acc[n], false, false);
      }
    }
  }

  // root part: K = 1024..1151 from hroot (bf16, row-major 128 wide)
  const __bf16* hrow = hroot + (size_t)row * kH + (hi ? 16 : 0);
#pragma unroll 1
  for (int kb = 0; kb < kH; kb += 32) {
    v8bf a0 = *(const v8bf*)(hrow + kb);
    v8bf a1 = *(const v8bf*)(hrow + kb + 8);
    v16bf a;
#pragma unroll
    for (int i = 0; i < 8; ++i) { a[i] = a0[i]; a[8 + i] = a1[i]; }
#pragma unroll
    for (int n = 0; n < 8; ++n) {
      v16bf b = *(const v16bf*)(wbase + (size_t)n * 16 * kKtot + kKagg + kb);
      acc[n] = __builtin_amdgcn_wmma_f32_16x16x32_bf16(false, a, false, b,
                                                       (short)0, acc[n], false, false);
    }
  }

  int rbase = m0 + (hi ? 8 : 0);
#pragma unroll
  for (int n = 0; n < 8; ++n) {
    int col = n * 16 + l;
    float bb = bias[col];
#pragma unroll
    for (int v = 0; v < 8; ++v) {
      float val = fmaxf(acc[n][v] + bb, 0.f);
      if (OUT_BF16) obf[(size_t)(rbase + v) * kH + col] = (__bf16)val;
      else          of32[(size_t)(rbase + v) * kH + col] = val;
    }
  }
}

// out[p] = h2[p] @ cls_w + cls_b : one wave per repo row, shuffle reduction
__global__ void k_cls(const float* __restrict__ h2, const float* __restrict__ cw,
                      const float* __restrict__ cb, float* __restrict__ out) {
  int wv = threadIdx.x >> 5, lane = threadIdx.x & 31;
  int p = blockIdx.x * 8 + wv;
  if (p >= kRepos) return;
  const float* r = h2 + (size_t)p * kH;
  float s0 = 0.f, s1 = 0.f;
#pragma unroll
  for (int k = lane; k < kH; k += 32) {
    float v = r[k];
    s0 += v * cw[k * 2 + 0];
    s1 += v * cw[k * 2 + 1];
  }
#pragma unroll
  for (int off = 16; off > 0; off >>= 1) {
    s0 += __shfl_down(s0, off, 32);
    s1 += __shfl_down(s1, off, 32);
  }
  if (lane == 0) {
    out[p * 2 + 0] = s0 + cb[0];
    out[p * 2 + 1] = s1 + cb[1];
  }
}

// ---------------- launch ----------------
extern "C" void kernel_launch(void* const* d_in, const int* in_sizes, int n_in,
                              void* d_out, int out_size, void* d_ws, size_t ws_size,
                              hipStream_t stream) {
  const float* x          = (const float*)d_in[0];
  const int*   edge_repo  = (const int*)d_in[1];
  const int*   edge_actor = (const int*)d_in[2];
  const int*   edge_type  = (const int*)d_in[3];
  const float* mlp_w      = (const float*)d_in[4];
  const float* mlp_b      = (const float*)d_in[5];
  const float* w1         = (const float*)d_in[6];
  const float* root1      = (const float*)d_in[7];
  const float* b1         = (const float*)d_in[8];
  const float* w2         = (const float*)d_in[9];
  const float* root2      = (const float*)d_in[10];
  const float* b2         = (const float*)d_in[11];
  const float* cls_w      = (const float*)d_in[12];
  const float* cls_b      = (const float*)d_in[13];
  float* out = (float*)d_out;

  float*  wsf   = (float*)d_ws;
  float*  agg1  = wsf + AGG1;
  float*  cnt1  = wsf + CNT1;
  float*  agg2  = wsf + AGG2;
  float*  cnt2  = wsf + CNT2;
  float*  h2    = wsf + H2OFF;
  __bf16* bbase = (__bf16*)((char*)d_ws + F32_TOT * sizeof(float));
  __bf16* hb    = bbase + HB;
  __bf16* h1b   = bbase + H1B;
  __bf16* wt1   = bbase + WT1;
  __bf16* wt2   = bbase + WT2;

  // 1. zero accumulators + counts (deterministic per call)
  {
    size_t n4 = ZERO_TOT / 4;
    int blocks = (int)((n4 + 255) / 256);
    k_zero<<<blocks, 256, 0, stream>>>(wsf, n4);
  }
  // 2. weights -> bf16, transposed to (N,K) with K contiguous
  k_prep_w<<<(2 * kH * kKtot + 255) / 256, 256, 0, stream>>>(w1, root1, w2, root2, wt1, wt2);
  // 3. input MLP
  k_mlp<<<(kN * kH) / 256, 256, 0, stream>>>(x, mlp_w, mlp_b, hb);
  // 4. layer-1 scatter: repo -> actor buckets
  k_scatter<<<kE / 8, 256, 0, stream>>>(hb, edge_repo, edge_actor, edge_type, 0, agg1, cnt1);
  // 5. layer-1 fused GEMM over all N nodes (repo tiles have no agg part)
  {
    int tiles = kN / 16;                       // 6250
    k_gemm<true><<<(tiles + 7) / 8, 256, 0, stream>>>(agg1, cnt1, hb, wt1, b1, h1b,
                                                      nullptr, tiles, kRepos);
  }
  // 6. layer-2 scatter: actor -> repo buckets
  k_scatter<<<kE / 8, 256, 0, stream>>>(h1b, edge_actor, edge_repo, edge_type, kRepos, agg2, cnt2);
  // 7. layer-2 fused GEMM over repos only (f32 out)
  {
    int tiles = kRepos / 16;                   // 1875
    k_gemm<false><<<(tiles + 7) / 8, 256, 0, stream>>>(agg2, cnt2, h1b, wt2, b2, nullptr,
                                                       h2, tiles, 0);
  }
  // 8. classifier
  k_cls<<<kRepos / 8, 256, 0, stream>>>(h2, cls_w, cls_b, out);
}
